// Wavelet_5059471475325
// MI455X (gfx1250) — matrix operand order repaired and checked by reading, not compiled, not run on hardware
//
#include <hip/hip_runtime.h>

#define NN   50000
#define EE   800000
#define HH   128
#define DIN  128
#define OUTD 64
#define GG   64

typedef __attribute__((ext_vector_type(16))) __bf16 v16bf;
typedef __attribute__((ext_vector_type(8)))  __bf16 v8bf;
typedef __attribute__((ext_vector_type(8)))  float  v8f;

// ---------------- utility kernels ----------------
__global__ void k_zero_i(int* p, int n){
    int i = blockIdx.x * blockDim.x + threadIdx.x;
    if (i < n) p[i] = 0;
}
__global__ void k_zero_f(float* p, int n){
    int i = blockIdx.x * blockDim.x + threadIdx.x;
    if (i < n) p[i] = 0.0f;
}

// in-degree count over destinations
__global__ void k_count(const int* __restrict__ ei, int* __restrict__ counts){
    int e = blockIdx.x * blockDim.x + threadIdx.x;
    if (e < EE) atomicAdd(&counts[ei[EE + e]], 1);
}

// single-workgroup exclusive scan over counts -> rowptr/cursor, plus deg_inv
__global__ void k_scan(const int* __restrict__ counts, int* __restrict__ rowptr,
                       int* __restrict__ cursor, float* __restrict__ deginv, int n){
    __shared__ int sd[1024];
    __shared__ int carry;
    if (threadIdx.x == 0) carry = 0;
    __syncthreads();
    for (int base = 0; base < n; base += 1024){
        int i = base + (int)threadIdx.x;
        int v = (i < n) ? counts[i] : 0;
        sd[threadIdx.x] = v;
        __syncthreads();
        for (int off = 1; off < 1024; off <<= 1){
            int t = (threadIdx.x >= (unsigned)off) ? sd[threadIdx.x - off] : 0;
            __syncthreads();
            sd[threadIdx.x] += t;
            __syncthreads();
        }
        int excl = sd[threadIdx.x] - v;
        if (i < n){
            int rp = carry + excl;
            rowptr[i] = rp;
            cursor[i] = rp;
            deginv[i] = (v > 0) ? 1.0f / (float)v : 0.0f;
        }
        int tot = sd[1023];
        __syncthreads();
        if (threadIdx.x == 0) carry += tot;
        __syncthreads();
    }
    if (threadIdx.x == 0) rowptr[n] = carry;
}

// scatter src indices into CSR-by-destination
__global__ void k_scatter(const int* __restrict__ ei, int* __restrict__ cursor,
                          int* __restrict__ csr){
    int e = blockIdx.x * blockDim.x + threadIdx.x;
    if (e >= EE) return;
    int dst = ei[EE + e];
    int src = ei[e];
    int pos = atomicAdd(&cursor[dst], 1);
    csr[pos] = src;
}

// W[k][n] fp32 -> Wt[n][k] bf16 (column-major weights for B fragments)
__global__ void k_transpose(const float* __restrict__ W, __bf16* __restrict__ Wt,
                            int K, int Ncol){
    int idx = blockIdx.x * blockDim.x + threadIdx.x;
    if (idx >= K * Ncol) return;
    int k = idx / Ncol, n = idx % Ncol;
    Wt[(size_t)n * K + k] = (__bf16)W[(size_t)k * Ncol + n];
}

// fused lazy-random-walk step: pout = 0.5*(pin + deg_inv * sum_{src in row} pin[src])
// one wave per node, 4 channels per lane (float4). CSR index broadcast via ds_bpermute.
__global__ void k_pstep(const float* __restrict__ pin, float* __restrict__ pout,
                        const int* __restrict__ rowptr, const int* __restrict__ csr,
                        const float* __restrict__ deginv){
    int node = blockIdx.x * 8 + (threadIdx.x >> 5);
    if (node >= NN) return;
    int lane = threadIdx.x & 31;
    int s = rowptr[node], e = rowptr[node + 1];
    float4 acc = make_float4(0.f, 0.f, 0.f, 0.f);
    int cbase = lane * 4;
    for (int b = s; b < e; b += 32){
        int myIdx = (b + lane < e) ? csr[b + lane] : 0;
        int cnt = min(32, e - b);
        for (int j = 0; j < cnt; ++j){
            int src = __builtin_amdgcn_ds_bpermute(j << 2, myIdx);
            const float4 v = *(const float4*)(pin + (size_t)src * HH + cbase);
            acc.x += v.x; acc.y += v.y; acc.z += v.z; acc.w += v.w;
        }
    }
    float di = deginv[node];
    const float4 pv = *(const float4*)(pin + (size_t)node * HH + cbase);
    float4 o;
    o.x = 0.5f * (pv.x + acc.x * di);
    o.y = 0.5f * (pv.y + acc.y * di);
    o.z = 0.5f * (pv.z + acc.z * di);
    o.w = 0.5f * (pv.w + acc.w * di);
    *(float4*)(pout + (size_t)node * HH + cbase) = o;
}

// segment-mean pooling (accumulate phase)
__global__ void k_pool(const float* __restrict__ h, const int* __restrict__ batch,
                       float* __restrict__ pooled, float* __restrict__ gcnt){
    int node = blockIdx.x * 8 + (threadIdx.x >> 5);
    if (node >= NN) return;
    int lane = threadIdx.x & 31;
    int g = batch[node];
    const float4 v = *(const float4*)(h + (size_t)node * HH + lane * 4);
    float* pp = pooled + (size_t)g * HH + lane * 4;
    atomicAdd(pp + 0, v.x); atomicAdd(pp + 1, v.y);
    atomicAdd(pp + 2, v.z); atomicAdd(pp + 3, v.w);
    if (lane == 0) atomicAdd(&gcnt[g], 1.0f);
}
__global__ void k_pool_fin(float* __restrict__ pooled, const float* __restrict__ gcnt){
    int i = blockIdx.x * blockDim.x + threadIdx.x;
    if (i < GG * HH) pooled[i] = pooled[i] / fmaxf(gcnt[i / HH], 1.0f);
}

// ---------------- WMMA GEMM ----------------
// A-fragment (16x32 bf16) per CDNA5 layout: lane holds row (lane&15);
// elements 0..7 = K offsets {0..7} + 8*(lane>=16), elements 8..15 = +16.
template<int AMODE>
__device__ inline v16bf make_afrag(const float* p1, const float* p2){
    float4 x0 = *(const float4*)(p1 + 0);
    float4 x1 = *(const float4*)(p1 + 4);
    float4 x2 = *(const float4*)(p1 + 16);
    float4 x3 = *(const float4*)(p1 + 20);
    if (AMODE == 2){
        float4 y0 = *(const float4*)(p2 + 0);
        float4 y1 = *(const float4*)(p2 + 4);
        float4 y2 = *(const float4*)(p2 + 16);
        float4 y3 = *(const float4*)(p2 + 20);
        x0.x -= y0.x; x0.y -= y0.y; x0.z -= y0.z; x0.w -= y0.w;
        x1.x -= y1.x; x1.y -= y1.y; x1.z -= y1.z; x1.w -= y1.w;
        x2.x -= y2.x; x2.y -= y2.y; x2.z -= y2.z; x2.w -= y2.w;
        x3.x -= y3.x; x3.y -= y3.y; x3.z -= y3.z; x3.w -= y3.w;
    }
    float f[16] = {x0.x,x0.y,x0.z,x0.w, x1.x,x1.y,x1.z,x1.w,
                   x2.x,x2.y,x2.z,x2.w, x3.x,x3.y,x3.z,x3.w};
    v16bf a;
#pragma unroll
    for (int i = 0; i < 16; ++i){
        float t = f[i];
        if (AMODE >= 1) t = fmaxf(t, 0.0f);
        a[i] = (__bf16)t;
    }
    return a;
}

// C[M,NCOL] = op(A[M,128]) @ W (+bias | +=C). W pre-transposed bf16 [NCOL][ldw], K=128.
// Register-blocked: one wave owns a 16-row strip and all NCOL/16 column tiles. Per
// k-step the A fragment is loaded + converted once, all NT B fragments are staged
// first (clause-batched loads), then NT WMMAs issue back-to-back into disjoint
// accumulators (no D->A/B hazards inside the chain).
// AMODE: 0 = A, 1 = relu(A), 2 = relu(A1 - A2)
template<int AMODE, bool ACCUM, int NCOL>
__global__ void k_gemm(const float* __restrict__ A1, const float* __restrict__ A2,
                       const __bf16* __restrict__ Wt, int ldw, int kw0,
                       const float* __restrict__ bias,
                       float* __restrict__ C, int M){
    constexpr int NT = NCOL / 16;            // column tiles per strip
    const int lane  = threadIdx.x & 31;
    const int wid   = threadIdx.x >> 5;
    const int strip = blockIdx.x * 8 + wid;
    if (strip >= (M >> 4)) return;           // uniform per wave; EXEC stays all-1s
    const int rt   = strip * 16;
    const int half = lane >> 4;
    const int r    = rt + (lane & 15);
    const int rb   = rt + half * 8;
    const int colb = lane & 15;

    // per-lane base of the lane's weight column block (row of Wt)
    const __bf16* wbase = Wt + (size_t)colb * ldw + kw0;

    v8f c[NT];
    if (ACCUM){
#pragma unroll
        for (int t = 0; t < NT; ++t)
#pragma unroll
            for (int v = 0; v < 8; ++v)
                c[t][v] = C[(size_t)(rb + v) * NCOL + t * 16 + colb];
    } else if (bias){
#pragma unroll
        for (int t = 0; t < NT; ++t){
            float bv = bias[t * 16 + colb];
#pragma unroll
            for (int v = 0; v < 8; ++v) c[t][v] = bv;
        }
    } else {
#pragma unroll
        for (int t = 0; t < NT; ++t)
#pragma unroll
            for (int v = 0; v < 8; ++v) c[t][v] = 0.0f;
    }

#pragma unroll
    for (int k0 = 0; k0 < HH; k0 += 32){
        // stage all NT B fragments first so the loads clause-batch and overlap
        v16bf bfr[NT];
#pragma unroll
        for (int t = 0; t < NT; ++t){
            const __bf16* pw = wbase + (size_t)t * 16 * ldw + k0;
            v8bf blo = *(const v8bf*)(pw + half * 8);
            v8bf bhi = *(const v8bf*)(pw + 16 + half * 8);
#pragma unroll
            for (int i = 0; i < 8; ++i){ bfr[t][i] = blo[i]; bfr[t][i + 8] = bhi[i]; }
        }

        const float* pa = A1 + (size_t)r * HH + k0 + half * 8;
        const float* pb = (AMODE == 2) ? (A2 + (size_t)r * HH + k0 + half * 8) : nullptr;
        v16bf a = make_afrag<AMODE>(pa, pb);   // loaded + converted once per k-step

#pragma unroll
        for (int t = 0; t < NT; ++t)
            c[t] = __builtin_amdgcn_wmma_f32_16x16x32_bf16(false, a, false, bfr[t],
                                                           (short)0, c[t], false, false);
    }
#pragma unroll
    for (int t = 0; t < NT; ++t)
#pragma unroll
        for (int v = 0; v < 8; ++v)
            C[(size_t)(rb + v) * NCOL + t * 16 + colb] = c[t][v];
}

// ---------------- host orchestration ----------------
extern "C" void kernel_launch(void* const* d_in, const int* in_sizes, int n_in,
                              void* d_out, int out_size, void* d_ws, size_t ws_size,
                              hipStream_t stream){
    const float* x     = (const float*)d_in[0];
    const int*   ei    = (const int*)  d_in[1];
    const int*   batch = (const int*)  d_in[2];
    const float* w_in  = (const float*)d_in[3];
    const float* b_in  = (const float*)d_in[4];
    const float* w1s[2] = {(const float*)d_in[5], (const float*)d_in[9]};
    const float* b1s[2] = {(const float*)d_in[6], (const float*)d_in[10]};
    const float* w2s[2] = {(const float*)d_in[7], (const float*)d_in[11]};
    const float* b2s[2] = {(const float*)d_in[8], (const float*)d_in[12]};
    const float* w_out = (const float*)d_in[13];
    const float* b_out = (const float*)d_in[14];
    float* out = (float*)d_out;

    char* ws = (char*)d_ws;
    size_t off = 0;
    auto alloc = [&](size_t bytes) -> char* {
        char* p = ws + off;
        off = (off + bytes + 255) & ~(size_t)255;
        return p;
    };
    const size_t NH = (size_t)NN * HH;
    float* hbuf  = (float*)alloc(NH * 4);
    float* bufA  = (float*)alloc(NH * 4);
    float* bufB  = (float*)alloc(NH * 4);
    float* cpbuf = (float*)alloc(NH * 4);
    float* acc   = (float*)alloc(NH * 4);
    float* pooled = (float*)alloc((size_t)GG * HH * 4);
    float* gcnt   = (float*)alloc((size_t)GG * 4);
    int*   counts = (int*)alloc((size_t)NN * 4);
    int*   rowptr = (int*)alloc((size_t)(NN + 1) * 4);
    int*   cursor = (int*)alloc((size_t)NN * 4);
    float* deginv = (float*)alloc((size_t)NN * 4);
    int*   csr    = (int*)alloc((size_t)EE * 4);
    __bf16* wint  = (__bf16*)alloc((size_t)DIN * HH * 2);
    __bf16* w1t[2] = {(__bf16*)alloc((size_t)6 * HH * HH * 2),
                      (__bf16*)alloc((size_t)6 * HH * HH * 2)};
    __bf16* w2t[2] = {(__bf16*)alloc((size_t)HH * HH * 2),
                      (__bf16*)alloc((size_t)HH * HH * 2)};
    __bf16* woutt = (__bf16*)alloc((size_t)HH * OUTD * 2);

    // ---- CSR build ----
    k_zero_i<<<(NN + 255) / 256, 256, 0, stream>>>(counts, NN);
    k_count<<<(EE + 255) / 256, 256, 0, stream>>>(ei, counts);
    k_scan<<<1, 1024, 0, stream>>>(counts, rowptr, cursor, deginv, NN);
    k_scatter<<<(EE + 255) / 256, 256, 0, stream>>>(ei, cursor, csr);

    // ---- weight transpose + bf16 convert ----
    auto tr = [&](const float* W, __bf16* Wt, int K, int Ncol){
        k_transpose<<<(K * Ncol + 255) / 256, 256, 0, stream>>>(W, Wt, K, Ncol);
    };
    tr(w_in, wint, DIN, HH);
    tr(w1s[0], w1t[0], 6 * HH, HH);
    tr(w2s[0], w2t[0], HH, HH);
    tr(w1s[1], w1t[1], 6 * HH, HH);
    tr(w2s[1], w2t[1], HH, HH);
    tr(w_out, woutt, HH, OUTD);

    const int strips = NN / 16;               // 3125 strip-waves per GEMM
    const int gN = (strips + 7) / 8;          // 391 blocks of 8 waves
    const int pB = (NN + 7) / 8;              // 6250 blocks for node-wave kernels

    auto pstep = [&](const float* in, float* o){
        k_pstep<<<pB, 256, 0, stream>>>(in, o, rowptr, csr, deginv);
    };

    // ---- input projection: h = x @ w_in + b_in ----
    k_gemm<0, false, HH><<<gN, 256, 0, stream>>>(x, nullptr, wint, DIN, 0, b_in, hbuf, NN);

    // ---- two wavelet layers ----
    for (int L = 0; L < 2; ++L){
        const __bf16* W1 = w1t[L];
        const __bf16* W2 = w2t[L];
        const float*  B1 = b1s[L];
        const float*  B2 = b2s[L];

        // t=1 : acc = relu(h - p1) @ W1[0] + b1
        pstep(hbuf, bufA);
        k_gemm<2, false, HH><<<gN, 256, 0, stream>>>(hbuf, bufA, W1, 6 * HH, 0 * HH, B1, acc, NN);
        // t=2 : acc += relu(p1 - p2) @ W1[1]
        pstep(bufA, bufB);
        k_gemm<2, true, HH><<<gN, 256, 0, stream>>>(bufA, bufB, W1, 6 * HH, 1 * HH, nullptr, acc, NN);
        hipMemcpyAsync(cpbuf, bufB, NH * 4, hipMemcpyDeviceToDevice, stream);
        // t=3,4 : acc += relu(p2 - p4) @ W1[2]
        pstep(bufB, bufA); pstep(bufA, bufB);
        k_gemm<2, true, HH><<<gN, 256, 0, stream>>>(cpbuf, bufB, W1, 6 * HH, 2 * HH, nullptr, acc, NN);
        hipMemcpyAsync(cpbuf, bufB, NH * 4, hipMemcpyDeviceToDevice, stream);
        // t=5..8 : acc += relu(p4 - p8) @ W1[3]
        pstep(bufB, bufA); pstep(bufA, bufB); pstep(bufB, bufA); pstep(bufA, bufB);
        k_gemm<2, true, HH><<<gN, 256, 0, stream>>>(cpbuf, bufB, W1, 6 * HH, 3 * HH, nullptr, acc, NN);
        hipMemcpyAsync(cpbuf, bufB, NH * 4, hipMemcpyDeviceToDevice, stream);
        // t=9..16 : acc += relu(p8 - p16) @ W1[4]
        pstep(bufB, bufA); pstep(bufA, bufB); pstep(bufB, bufA); pstep(bufA, bufB);
        pstep(bufB, bufA); pstep(bufA, bufB); pstep(bufB, bufA); pstep(bufA, bufB);
        k_gemm<2, true, HH><<<gN, 256, 0, stream>>>(cpbuf, bufB, W1, 6 * HH, 4 * HH, nullptr, acc, NN);
        // last block: acc += relu(p16) @ W1[5]
        k_gemm<1, true, HH><<<gN, 256, 0, stream>>>(bufB, nullptr, W1, 6 * HH, 5 * HH, nullptr, acc, NN);
        // h = relu(acc) @ W2 + b2
        k_gemm<1, false, HH><<<gN, 256, 0, stream>>>(acc, nullptr, W2, HH, 0, B2, hbuf, NN);
    }

    // ---- segment-mean pooling + output head ----
    k_zero_f<<<(GG * HH + 255) / 256, 256, 0, stream>>>(pooled, GG * HH);
    k_zero_f<<<1, 256, 0, stream>>>(gcnt, GG);
    k_pool<<<pB, 256, 0, stream>>>(hbuf, batch, pooled, gcnt);
    k_pool_fin<<<(GG * HH + 255) / 256, 256, 0, stream>>>(pooled, gcnt);

    // final head: pooled[64,128] @ w_out -> out[64,64]; 4 strip-waves in 1 block
    k_gemm<0, false, OUTD><<<1, 256, 0, stream>>>(pooled, nullptr, woutt, HH, 0,
                                                  b_out, out, GG);
}
